// SVDHead_2980707304114
// MI455X (gfx1250) — matrix operand order repaired and checked by reading, not compiled, not run on hardware
//
#include <hip/hip_runtime.h>
#include <math.h>
#include <stdint.h>

typedef __attribute__((ext_vector_type(16))) __bf16 v16bf;
typedef __attribute__((ext_vector_type(8)))  float  v8f;

static constexpr int Bn = 8, Dk = 512, Nn = 2048, Mm = 2048;
static constexpr int ROWS_PER_BLOCK = 128;        // 8 waves * 16 rows
static constexpr int MT = 64;                     // m-tile (columns of scores)
static constexpr int KSTRIDE = Dk + 8;            // bf16 elems; 1040B row, 4-bank skew
static constexpr unsigned KROWB = KSTRIDE * 2;    // 1040 bytes (both global K and LDS)
static constexpr unsigned LDSBUF = (unsigned)MT * KROWB;   // 66,560 B per buffer

// ---------------------------------------------------------------------------
// Kernel 1: [B, D, Ncols] f32 -> [B, Ncols, ostride] bf16 (scaled, RNE)
// ---------------------------------------------------------------------------
__global__ void cvt_transpose(const float* __restrict__ in,
                              unsigned short* __restrict__ out,
                              int Drows, int Ncols, int ostride, float scale) {
  __shared__ float tile[32][33];
  const int b = blockIdx.z;
  const int n0 = blockIdx.x * 32, d0 = blockIdx.y * 32;
  const int tx = threadIdx.x, ty = threadIdx.y;   // (32, 8)
  const float* ip = in + (size_t)b * Drows * Ncols;
  unsigned short* op = out + (size_t)b * Ncols * ostride;
#pragma unroll
  for (int i = 0; i < 4; ++i) {
    int d = d0 + ty + i * 8;
    tile[ty + i * 8][tx] = ip[(size_t)d * Ncols + (n0 + tx)];
  }
  __syncthreads();
#pragma unroll
  for (int i = 0; i < 4; ++i) {
    int n = n0 + ty + i * 8;
    float f = tile[tx][ty + i * 8] * scale;
    unsigned u = __float_as_uint(f);
    u += 0x7FFFu + ((u >> 16) & 1u);              // round-to-nearest-even
    op[(size_t)n * ostride + (d0 + tx)] = (unsigned short)(u >> 16);
  }
}

// ---------------------------------------------------------------------------
// Kernel 2: fused attention -> src_corr[B,3,N]
//   Q resident in registers; K tiles DMA'd global->LDS with
//   GLOBAL_LOAD_ASYNC_TO_LDS_B128 (ASYNCcnt), double-buffered.
//   Global K layout is padded to the same 1040B row stride as LDS, so one
//   VGPR offset serves both addresses and chunks use immediate offsets only
//   (IOFFSET is added to BOTH lds and global addresses per ISA 15.18).
// ---------------------------------------------------------------------------
union FragU { uint4 u4[2]; v16bf v; };

#define ALOAD(OFF)                                                      \
  asm volatile("global_load_async_to_lds_b128 %0, %1, %2 offset:" #OFF \
               :: "v"(ldsoff), "v"(goff), "s"(gbase) : "memory")

__device__ __forceinline__ void stage_tile(uint64_t gbase, unsigned goff,
                                           unsigned ldsoff) {
  ALOAD(0);     ALOAD(4160);  ALOAD(8320);  ALOAD(12480);
  ALOAD(16640); ALOAD(20800); ALOAD(24960); ALOAD(29120);
  ALOAD(33280); ALOAD(37440); ALOAD(41600); ALOAD(45760);
  ALOAD(49920); ALOAD(54080); ALOAD(58240); ALOAD(62400);
}

__global__ void __launch_bounds__(256) attn_corr(
    const unsigned short* __restrict__ Qb,   // [B,N,512] bf16 (pre-scaled)
    const unsigned short* __restrict__ Kb,   // [B,M,520] bf16 (padded rows)
    const float* __restrict__ tgt,           // [B,3,M] f32
    float* __restrict__ corr)                // [B,3,N] f32
{
  extern __shared__ __align__(16) char smem[];
  const int b     = blockIdx.y;
  const int nbase = blockIdx.x * ROWS_PER_BLOCK;
  const int tid   = threadIdx.x;
  const int wave  = tid >> 5, lane = tid & 31;
  const int r = lane & 15, hi = lane >> 4;

  // ---- load resident Q fragments (ISA 16-bit A 16x32 layout) ----
  FragU qf[16];
  {
    const unsigned short* qrow =
        Qb + ((size_t)b * Nn + nbase + wave * 16 + r) * Dk;
#pragma unroll
    for (int f = 0; f < 16; ++f) {
      int d0 = f * 32 + hi * 8;
      qf[f].u4[0] = *reinterpret_cast<const uint4*>(qrow + d0);
      qf[f].u4[1] = *reinterpret_cast<const uint4*>(qrow + d0 + 16);
    }
  }

  float mrow[8], lrow[8], acc0[8], acc1[8], acc2[8];
#pragma unroll
  for (int v = 0; v < 8; ++v) {
    mrow[v] = -1e30f; lrow[v] = 0.f;
    acc0[v] = acc1[v] = acc2[v] = 0.f;
  }

  // shared byte offset used for BOTH global and LDS addressing
  const unsigned goff    = ((unsigned)(tid >> 6)) * KROWB + ((unsigned)(tid & 63)) * 16u;
  const unsigned ldsbase = (unsigned)(uintptr_t)(void*)smem;
  const uint64_t kb64    = (uint64_t)(uintptr_t)(Kb + (size_t)b * Mm * KSTRIDE);

  // prologue: DMA tile 0 into buffer 0
  stage_tile(kb64, goff, ldsbase + goff);

  for (int m0 = 0; m0 < Mm; m0 += MT) {
    const unsigned par = (unsigned)((m0 / MT) & 1);
    // wait for this tile's async DMA, then make it visible to all waves
    asm volatile("s_wait_asynccnt 0x0" ::: "memory");
    __syncthreads();
    // prefetch next tile into the other buffer (its readers are all past
    // the barrier above, so the overwrite is safe)
    if (m0 + MT < Mm)
      stage_tile(kb64 + (uint64_t)(m0 + MT) * KROWB, goff,
                 ldsbase + (par ^ 1u) * LDSBUF + goff);

    const char* buf = smem + par * LDSBUF;

    // ---- scores: 4 sub-tiles of 16 columns, 16 WMMAs each ----
    v8f S[4];
#pragma unroll
    for (int sub = 0; sub < 4; ++sub) {
      v8f s = {0.f,0.f,0.f,0.f,0.f,0.f,0.f,0.f};
      const char* kcol = buf + (unsigned)(sub * 16 + r) * KROWB + hi * 16;
#pragma unroll
      for (int f = 0; f < 16; ++f) {
        FragU kb;
        kb.u4[0] = *reinterpret_cast<const uint4*>(kcol + f * 64);
        kb.u4[1] = *reinterpret_cast<const uint4*>(kcol + f * 64 + 32);
        s = __builtin_amdgcn_wmma_f32_16x16x32_bf16(false, qf[f].v,
                                                    false, kb.v,
                                                    (short)0, s, false, false);
      }
      S[sub] = s;
    }

    // ---- per-tile tgt (V) values: column m0+sub*16+r ----
    float tv0[4], tv1[4], tv2[4];
#pragma unroll
    for (int sub = 0; sub < 4; ++sub) {
      int m = m0 + sub * 16 + r;
      tv0[sub] = tgt[((size_t)b * 3 + 0) * Mm + m];
      tv1[sub] = tgt[((size_t)b * 3 + 1) * Mm + m];
      tv2[sub] = tgt[((size_t)b * 3 + 2) * Mm + m];
    }

    // ---- online softmax + V accumulation (row v+8*hi per VGPR slot v) ----
#pragma unroll
    for (int v = 0; v < 8; ++v) {
      float mx = fmaxf(fmaxf(S[0][v], S[1][v]), fmaxf(S[2][v], S[3][v]));
#pragma unroll
      for (int off = 1; off <= 8; off <<= 1)
        mx = fmaxf(mx, __shfl_xor(mx, off, 32));
      float mnew = fmaxf(mrow[v], mx);
      float sc = __builtin_amdgcn_exp2f(mrow[v] - mnew);
      float p0 = __builtin_amdgcn_exp2f(S[0][v] - mnew);
      float p1 = __builtin_amdgcn_exp2f(S[1][v] - mnew);
      float p2 = __builtin_amdgcn_exp2f(S[2][v] - mnew);
      float p3 = __builtin_amdgcn_exp2f(S[3][v] - mnew);
      lrow[v] = lrow[v] * sc + (p0 + p1 + p2 + p3);
      acc0[v] = acc0[v] * sc + (p0*tv0[0] + p1*tv0[1] + p2*tv0[2] + p3*tv0[3]);
      acc1[v] = acc1[v] * sc + (p0*tv1[0] + p1*tv1[1] + p2*tv1[2] + p3*tv1[3]);
      acc2[v] = acc2[v] * sc + (p0*tv2[0] + p1*tv2[1] + p2*tv2[2] + p3*tv2[3]);
      mrow[v] = mnew;
    }
  }

  // ---- reduce lane-partials across the 16-lane group, write src_corr ----
#pragma unroll
  for (int v = 0; v < 8; ++v) {
    float l = lrow[v], a0 = acc0[v], a1 = acc1[v], a2 = acc2[v];
#pragma unroll
    for (int off = 1; off <= 8; off <<= 1) {
      l  += __shfl_xor(l,  off, 32);
      a0 += __shfl_xor(a0, off, 32);
      a1 += __shfl_xor(a1, off, 32);
      a2 += __shfl_xor(a2, off, 32);
    }
    if (r == 0) {
      int row = nbase + wave * 16 + v + hi * 8;
      float inv = 1.f / l;
      corr[((size_t)b * 3 + 0) * Nn + row] = a0 * inv;
      corr[((size_t)b * 3 + 1) * Nn + row] = a1 * inv;
      corr[((size_t)b * 3 + 2) * Nn + row] = a2 * inv;
    }
  }
}

// ---------------------------------------------------------------------------
// Kernel 3: means, H = src_c * corr_c^T, Horn quaternion Procrustes, R & t
// ---------------------------------------------------------------------------
__global__ void __launch_bounds__(256) procrustes_finalize(
    const float* __restrict__ src, const float* __restrict__ tgt,
    const float* __restrict__ corr, float* __restrict__ out)
{
  __shared__ float tot[18];
  const int b = blockIdx.x, tid = threadIdx.x;
  if (tid < 18) tot[tid] = 0.f;
  __syncthreads();

  float ps[3] = {0,0,0}, pc[3] = {0,0,0}, pt[3] = {0,0,0};
  float pP[9] = {0,0,0,0,0,0,0,0,0};
  for (int n = tid; n < Nn; n += 256) {
    float sv[3], cv[3];
#pragma unroll
    for (int i = 0; i < 3; ++i) {
      sv[i] = src [((size_t)b * 3 + i) * Nn + n];
      cv[i] = corr[((size_t)b * 3 + i) * Nn + n];
      pt[i] += tgt[((size_t)b * 3 + i) * Mm + n];   // M == N
      ps[i] += sv[i];
      pc[i] += cv[i];
    }
#pragma unroll
    for (int i = 0; i < 3; ++i)
#pragma unroll
      for (int j = 0; j < 3; ++j) pP[i * 3 + j] += sv[i] * cv[j];
  }
#pragma unroll
  for (int k = 0; k < 3; ++k) {
    atomicAdd(&tot[k],     ps[k]);
    atomicAdd(&tot[3 + k], pc[k]);
    atomicAdd(&tot[6 + k], pt[k]);
  }
#pragma unroll
  for (int k = 0; k < 9; ++k) atomicAdd(&tot[9 + k], pP[k]);
  __syncthreads();

  if (tid == 0) {
    float ms[3], mc[3], mt[3], H[9];
    for (int i = 0; i < 3; ++i) {
      ms[i] = tot[i] / Nn; mc[i] = tot[3 + i] / Nn; mt[i] = tot[6 + i] / Mm;
    }
    for (int i = 0; i < 3; ++i)
      for (int j = 0; j < 3; ++j)
        H[i * 3 + j] = tot[9 + i * 3 + j] - (float)Nn * ms[i] * mc[j];

    // Horn's 4x4 N-matrix: max eigenvector quaternion == det-fixed V*U^T
    float Sxx=H[0],Sxy=H[1],Sxz=H[2],Syx=H[3],Syy=H[4],Syz=H[5],Szx=H[6],Szy=H[7],Szz=H[8];
    float K4[4][4] = {
      { Sxx+Syy+Szz, Syz-Szy,      Szx-Sxz,      Sxy-Syx      },
      { Syz-Szy,     Sxx-Syy-Szz,  Sxy+Syx,      Szx+Sxz      },
      { Szx-Sxz,     Sxy+Syx,     -Sxx+Syy-Szz,  Syz+Szy      },
      { Sxy-Syx,     Szx+Sxz,      Syz+Szy,     -Sxx-Syy+Szz  }};
    float shift = 0.f;
    for (int i = 0; i < 4; ++i) {
      float rs = fabsf(K4[i][0]) + fabsf(K4[i][1]) + fabsf(K4[i][2]) + fabsf(K4[i][3]);
      shift = fmaxf(shift, rs);
    }
    float q[4] = {1.f, 0.0123f, 0.0231f, 0.0312f};
    for (int it = 0; it < 96; ++it) {
      float nq[4];
      for (int i = 0; i < 4; ++i)
        nq[i] = shift * q[i] + K4[i][0]*q[0] + K4[i][1]*q[1] + K4[i][2]*q[2] + K4[i][3]*q[3];
      float inv = rsqrtf(nq[0]*nq[0] + nq[1]*nq[1] + nq[2]*nq[2] + nq[3]*nq[3]);
      for (int i = 0; i < 4; ++i) q[i] = nq[i] * inv;
    }
    float w = q[0], x = q[1], y = q[2], z = q[3];
    float R[9] = {
      1.f-2.f*(y*y+z*z), 2.f*(x*y-w*z),     2.f*(x*z+w*y),
      2.f*(x*y+w*z),     1.f-2.f*(x*x+z*z), 2.f*(y*z-w*x),
      2.f*(x*z-w*y),     2.f*(y*z+w*x),     1.f-2.f*(x*x+y*y)};
    for (int k = 0; k < 9; ++k) out[b * 9 + k] = R[k];
    for (int i = 0; i < 3; ++i)
      out[Bn * 9 + b * 3 + i] =
          mt[i] - (R[i*3+0]*ms[0] + R[i*3+1]*ms[1] + R[i*3+2]*ms[2]);
  }
}

// ---------------------------------------------------------------------------
extern "C" void kernel_launch(void* const* d_in, const int* in_sizes, int n_in,
                              void* d_out, int out_size, void* d_ws, size_t ws_size,
                              hipStream_t stream) {
  const float* srce = (const float*)d_in[0];  // [B,D,N]
  const float* tgte = (const float*)d_in[1];  // [B,D,M]
  const float* src  = (const float*)d_in[2];  // [B,3,N]
  const float* tgt  = (const float*)d_in[3];  // [B,3,M]
  float* out = (float*)d_out;

  unsigned short* Qb = (unsigned short*)d_ws;               // [B,N,512] bf16
  unsigned short* Kb = Qb + (size_t)Bn * Nn * Dk;           // [B,M,520] bf16 (padded)
  float* corr = (float*)((char*)Kb + (size_t)Bn * Mm * KSTRIDE * sizeof(unsigned short));

  // fold 1/sqrt(d_k) and log2(e) into Q so softmax uses exp2 directly
  const float qscale = 1.4426950408889634f / sqrtf((float)Dk);
  dim3 tb(32, 8);
  cvt_transpose<<<dim3(Nn / 32, Dk / 32, Bn), tb, 0, stream>>>(srce, Qb, Dk, Nn, Dk, qscale);
  cvt_transpose<<<dim3(Mm / 32, Dk / 32, Bn), tb, 0, stream>>>(tgte, Kb, Dk, Mm, KSTRIDE, 1.0f);

  attn_corr<<<dim3(Nn / ROWS_PER_BLOCK, Bn), 256, 2 * LDSBUF, stream>>>(Qb, Kb, tgt, corr);

  procrustes_finalize<<<dim3(Bn), 256, 0, stream>>>(src, tgt, corr, out);
}
